// BitLinearLRLS_37391985279696
// MI455X (gfx1250) — matrix operand order repaired and checked by reading, not compiled, not run on hardware
//
#include <hip/hip_runtime.h>
#include <hip/hip_bf16.h>

// Problem dims (compile-time constants from the reference)
#define IN_DIM   4096
#define OUT_DIM  4096
#define MROWS    16384   // B*S = 4*4096
#define RANK     16

typedef __attribute__((ext_vector_type(16))) __bf16 v16bf;
typedef __attribute__((ext_vector_type(8)))  float  v8f;

// Types matching the async-to-LDS builtin prototype: v4i32* in AS1/AS3
typedef int v4i __attribute__((vector_size(16)));
typedef __attribute__((address_space(1))) v4i* gv4i_p;   // global pointee
typedef __attribute__((address_space(3))) v4i* lv4i_p;   // LDS pointee

// gfx1250 async global->LDS path, guarded so the file compiles on host pass
// and on toolchains that do not expose the builtins.
#if defined(__AMDGCN__) && __has_builtin(__builtin_amdgcn_global_load_async_to_lds_b128) && __has_builtin(__builtin_amdgcn_s_wait_asynccnt)
#define USE_ASYNC 1
#endif

__device__ __forceinline__ void copy16(const unsigned short* src, unsigned short* dst) {
#ifdef USE_ASYNC
    __builtin_amdgcn_global_load_async_to_lds_b128((gv4i_p)src, (lv4i_p)dst, 0, 0);
#else
    *(uint4*)dst = *(const uint4*)src;
#endif
}

__device__ __forceinline__ void async_wait_all() {
#ifdef USE_ASYNC
    __builtin_amdgcn_s_wait_asynccnt(0);
#endif
}

__device__ __forceinline__ unsigned short f2bf(float f) {
    union { float f; unsigned u; } v; v.f = f;
    unsigned u = v.u;
    return (unsigned short)((u + 0x7FFFu + ((u >> 16) & 1u)) >> 16);  // RNE
}

// ---------------------------------------------------------------------------
// Prologue 1: partial column sums of x: partial[by][col] = sum of 256 rows
// ---------------------------------------------------------------------------
__global__ void __launch_bounds__(256) colsum_kernel(const float* __restrict__ x,
                                                     float* __restrict__ partial) {
    const int col = blockIdx.x * 256 + threadIdx.x;   // gridDim.x = 16
    const int r0  = blockIdx.y * 256;                 // gridDim.y = 64
    float s = 0.f;
    #pragma unroll 4
    for (int r = 0; r < 256; ++r)
        s += x[(size_t)(r0 + r) * IN_DIM + col];
    partial[(size_t)blockIdx.y * IN_DIM + col] = s;
}

__global__ void __launch_bounds__(256) colsum_reduce_kernel(const float* __restrict__ partial,
                                                            float* __restrict__ xsum) {
    const int col = blockIdx.x * 256 + threadIdx.x;   // gridDim.x = 16
    float s = 0.f;
    #pragma unroll 8
    for (int j = 0; j < 64; ++j)
        s += partial[(size_t)j * IN_DIM + col];
    xsum[col] = s;
}

// ---------------------------------------------------------------------------
// Prologue 2: scale_eff[o] = scale[o] + A[o,:] @ (Bm @ (xsum/16384))
// ---------------------------------------------------------------------------
__global__ void __launch_bounds__(256) scale_eff_kernel(const float* __restrict__ lrA,
                                                        const float* __restrict__ lrB,
                                                        const float* __restrict__ scale,
                                                        const float* __restrict__ xsum,
                                                        float* __restrict__ se) {
    __shared__ float part[RANK][17];
    __shared__ float t[RANK];
    const int tid = threadIdx.x;
    const int r = tid >> 4, c = tid & 15;
    float s = 0.f;
    for (int i = c; i < IN_DIM; i += 16)
        s += lrB[r * IN_DIM + i] * xsum[i];
    part[r][c] = s;
    __syncthreads();
    if (c == 0) {
        float tt = 0.f;
        #pragma unroll
        for (int j = 0; j < 16; ++j) tt += part[r][j];
        t[r] = tt * (1.0f / (float)MROWS);
    }
    __syncthreads();
    const int o = blockIdx.x * 256 + tid;             // gridDim.x = 16
    float corr = 0.f;
    #pragma unroll
    for (int rr = 0; rr < RANK; ++rr) corr += lrA[o * RANK + rr] * t[rr];
    se[o] = scale[o] + corr;
}

// ---------------------------------------------------------------------------
// Prologue 3: ternary-quantize weight -> bf16  (values in {-1,0,1}, exact)
// ---------------------------------------------------------------------------
__global__ void __launch_bounds__(256) quant_w_kernel(const float* __restrict__ w,
                                                      unsigned short* __restrict__ wq) {
    const size_t i = ((size_t)blockIdx.x * 256 + threadIdx.x) * 8;
    float4 a = *(const float4*)(w + i);
    float4 b = *(const float4*)(w + i + 4);
    unsigned short o[8];
    o[0] = f2bf(fminf(fmaxf(rintf(a.x * 2.0f), -1.f), 1.f));
    o[1] = f2bf(fminf(fmaxf(rintf(a.y * 2.0f), -1.f), 1.f));
    o[2] = f2bf(fminf(fmaxf(rintf(a.z * 2.0f), -1.f), 1.f));
    o[3] = f2bf(fminf(fmaxf(rintf(a.w * 2.0f), -1.f), 1.f));
    o[4] = f2bf(fminf(fmaxf(rintf(b.x * 2.0f), -1.f), 1.f));
    o[5] = f2bf(fminf(fmaxf(rintf(b.y * 2.0f), -1.f), 1.f));
    o[6] = f2bf(fminf(fmaxf(rintf(b.z * 2.0f), -1.f), 1.f));
    o[7] = f2bf(fminf(fmaxf(rintf(b.w * 2.0f), -1.f), 1.f));
    *(uint4*)(wq + i) = *(const uint4*)o;
}

// ---------------------------------------------------------------------------
// Prologue 4: convert x -> bf16
// ---------------------------------------------------------------------------
__global__ void __launch_bounds__(256) conv_x_kernel(const float* __restrict__ x,
                                                     unsigned short* __restrict__ xb) {
    const size_t i = ((size_t)blockIdx.x * 256 + threadIdx.x) * 8;
    float4 a = *(const float4*)(x + i);
    float4 b = *(const float4*)(x + i + 4);
    unsigned short o[8];
    o[0] = f2bf(a.x); o[1] = f2bf(a.y); o[2] = f2bf(a.z); o[3] = f2bf(a.w);
    o[4] = f2bf(b.x); o[5] = f2bf(b.y); o[6] = f2bf(b.z); o[7] = f2bf(b.w);
    *(uint4*)(xb + i) = *(const uint4*)o;
}

// ---------------------------------------------------------------------------
// Main GEMM: y[m,n] = se[n] * sum_k xb[m,k] * wq[n,k]
// block tile 128x128, BK=32, 8 waves (4x2), wave tile 32x64 (2x4 WMMA accs)
// ---------------------------------------------------------------------------
#define BK 32
#define LDS_STRIDE 40   // padded row stride (elements); 80 bytes, 16B aligned

__global__ void __launch_bounds__(256, 2)
gemm_bf16_wmma(const unsigned short* __restrict__ xb,
               const unsigned short* __restrict__ wq,
               const float* __restrict__ se,
               float* __restrict__ out) {
    __shared__ __align__(16) unsigned short As[2][128][LDS_STRIDE];
    __shared__ __align__(16) unsigned short Bs[2][128][LDS_STRIDE];

    const int tid    = threadIdx.x;
    const int lane   = tid & 31;
    const int wave   = tid >> 5;
    const int lane16 = lane & 15;
    const int kh     = lane >> 4;      // which half of the wave
    const int wm     = wave >> 1;      // 0..3  -> M offset wm*32
    const int wn     = wave & 1;       // 0..1  -> N offset wn*64
    const int bm     = blockIdx.y * 128;
    const int bn     = blockIdx.x * 128;

    v8f acc[2][4];
    #pragma unroll
    for (int i = 0; i < 2; ++i)
        #pragma unroll
        for (int j = 0; j < 4; ++j) {
            v8f z = {0.f, 0.f, 0.f, 0.f, 0.f, 0.f, 0.f, 0.f};
            acc[i][j] = z;
        }

    // Per-thread tile-staging coordinates (trip count was always 2: unrolled).
    // 512 16B-chunks per 128x32 tile; thread handles chunks tid and tid+256.
    const int crow = tid >> 2;          // 0..63
    const int ce   = (tid & 3) * 8;     // element offset within row
    const unsigned short* xsrc0 = xb + (size_t)(bm + crow)      * IN_DIM + ce;
    const unsigned short* xsrc1 = xb + (size_t)(bm + crow + 64) * IN_DIM + ce;
    const unsigned short* wsrc0 = wq + (size_t)(bn + crow)      * IN_DIM + ce;
    const unsigned short* wsrc1 = wq + (size_t)(bn + crow + 64) * IN_DIM + ce;

    auto load_tiles = [&](int buf, int k0) {
        copy16(xsrc0 + k0, &As[buf][crow][ce]);
        copy16(wsrc0 + k0, &Bs[buf][crow][ce]);
        copy16(xsrc1 + k0, &As[buf][crow + 64][ce]);
        copy16(wsrc1 + k0, &Bs[buf][crow + 64][ce]);
    };

    load_tiles(0, 0);
    async_wait_all();
    __syncthreads();

    const int KSTEPS = IN_DIM / BK;   // 128
    for (int kt = 0; kt < KSTEPS; ++kt) {
        const int cur = kt & 1;
        if (kt + 1 < KSTEPS) load_tiles(cur ^ 1, (kt + 1) * BK);

        union Frag { v16bf v; uint4 q[2]; };
        Frag a[2], b[4];
        // A 16x32 bf16 layout: lane L holds row M=L%16;
        // K = {kh*8 .. kh*8+7} U {16+kh*8 .. 16+kh*8+7}
        #pragma unroll
        for (int tm = 0; tm < 2; ++tm) {
            const unsigned short* p = &As[cur][wm * 32 + tm * 16 + lane16][kh * 8];
            a[tm].q[0] = *(const uint4*)p;
            a[tm].q[1] = *(const uint4*)(p + 16);
        }
        // B 32x16 bf16 layout: lane L holds column N=L%16; K = {kh*8 .. kh*8+7} U {16+kh*8 .. 16+kh*8+7}
        #pragma unroll
        for (int tn = 0; tn < 4; ++tn) {
            const unsigned short* p = &Bs[cur][wn * 64 + tn * 16 + lane16][kh * 8];
            b[tn].q[0] = *(const uint4*)p;
            b[tn].q[1] = *(const uint4*)(p + 16);
        }
        #pragma unroll
        for (int tm = 0; tm < 2; ++tm)
            #pragma unroll
            for (int tn = 0; tn < 4; ++tn)
                acc[tm][tn] = __builtin_amdgcn_wmma_f32_16x16x32_bf16(
                    false, a[tm].v, false, b[tn].v,
                    (short)0, acc[tm][tn], false, false);

        if (kt + 1 < KSTEPS) async_wait_all();
        __syncthreads();
    }

    // Epilogue: C layout (f32 16x16): VGPR r, lane L -> M = kh*8 + r, N = L%16
    #pragma unroll
    for (int tn = 0; tn < 4; ++tn) {
        const int n = bn + wn * 64 + tn * 16 + lane16;
        const float sv = se[n];
        #pragma unroll
        for (int tm = 0; tm < 2; ++tm) {
            const int mbase = bm + wm * 32 + tm * 16 + kh * 8;
            #pragma unroll
            for (int r = 0; r < 8; ++r)
                out[(size_t)(mbase + r) * OUT_DIM + n] = acc[tm][tn][r] * sv;
        }
    }
}

// ---------------------------------------------------------------------------
// Host launcher
// ---------------------------------------------------------------------------
extern "C" void kernel_launch(void* const* d_in, const int* in_sizes, int n_in,
                              void* d_out, int out_size, void* d_ws, size_t ws_size,
                              hipStream_t stream) {
    const float* x      = (const float*)d_in[0];   // [16384, 4096]
    const float* weight = (const float*)d_in[1];   // [4096, 4096]
    // d_in[2] = threshold (unused: reference uses constant THRESH=0.5)
    const float* scale  = (const float*)d_in[3];   // [4096]
    const float* lrA    = (const float*)d_in[4];   // [4096, 16]
    const float* lrB    = (const float*)d_in[5];   // [16, 4096]
    float* out = (float*)d_out;

    char* ws = (char*)d_ws;
    float*          xsum    = (float*)(ws);                         // 16 KB
    float*          se      = (float*)(ws + (16 << 10));            // 16 KB
    float*          partial = (float*)(ws + (32 << 10));            // 64*4096*4 = 1 MB
    unsigned short* wq      = (unsigned short*)(ws + (2ull << 20)); // 32 MB
    unsigned short* xb      = (unsigned short*)(ws + (2ull << 20) + (size_t)OUT_DIM * IN_DIM * 2); // 128 MB

    colsum_kernel<<<dim3(16, 64), 256, 0, stream>>>(x, partial);
    colsum_reduce_kernel<<<dim3(16), 256, 0, stream>>>(partial, xsum);
    scale_eff_kernel<<<dim3(16), 256, 0, stream>>>(lrA, lrB, scale, xsum, se);
    quant_w_kernel<<<dim3((OUT_DIM * IN_DIM) / (256 * 8)), 256, 0, stream>>>(weight, wq);
    conv_x_kernel<<<dim3(((size_t)MROWS * IN_DIM) / (256 * 8)), 256, 0, stream>>>(x, xb);
    gemm_bf16_wmma<<<dim3(OUT_DIM / 128, MROWS / 128), 256, 0, stream>>>(xb, wq, se, out);
}
